// Decoder_8452495638856
// MI455X (gfx1250) — matrix operand order repaired and checked by reading, not compiled, not run on hardware
//
#include <hip/hip_runtime.h>

// ============================================================================
// Tacotron2-style decoder for MI455X (gfx1250, wave32, WMMA).
//
// Strategy:
//  * One persistent kernel (32 WGs x 256 thr = 256 wave32 waves) runs the
//    whole 500-step recurrence; device-wide barrier = monotonic atomic
//    counter + s_sleep spin (3 syncs/step).
//  * LSTM GEMMs ([32 x K] @ [K x 4096]) use v_wmma_f32_16x16x32_f16 with
//    f32 accumulate. Weights are converted to f16 and pre-packed into
//    B-fragment tile order (1KB/tile, 32B/lane -> 2x global_load_b128),
//    ~36MB total => fully L2-resident on the 192MB L2.
//  * Activations [x|ctx|h] staged as f16 in LDS each step (<=160KB of the
//    320KB WGP LDS); A fragments = 2x ds_read_b128 per k-tile.
//  * Attention (conv31 + loc-dense + tanh energies + 2 softmaxes + forward
//    attention + context) runs one batch row per WG; alpha / cumulative
//    attention live in LDS for the entire kernel (no global round trips).
//  * Precompute kernels each call: prenet, pmem = memory@wmem.T, weight
//    packing, state zeroing. All deterministic, graph-capture safe.
// ============================================================================

typedef __attribute__((ext_vector_type(16))) _Float16 v16h;
typedef __attribute__((ext_vector_type(8)))  float    v8f;

static constexpr int kB    = 32;
static constexpr int kTenc = 400;
static constexpr int kTdec = 500;
static constexpr int kMel  = 80;
static constexpr int kPre  = 256;
static constexpr int kEnc  = 512;
static constexpr int kRnn  = 1024;   // ARNN == DRNN
static constexpr int kAdim = 128;
static constexpr int kNf   = 32;     // NFILT
static constexpr int kKA   = kPre + kEnc + kRnn;   // 1792: [pre|ctx|h]
static constexpr int kKD   = kRnn + kEnc + kRnn;   // 2560: [ah|ctx|dh]
static constexpr int kKtA  = kKA / 32;             // 56 k-tiles
static constexpr int kKtD  = kKD / 32;             // 80 k-tiles
static constexpr int kNWG  = 32;
static constexpr int kThr  = 256;

// ---------------- workspace layout (bytes) ----------------
static constexpr size_t OFF_CNT  = 0;                                   // 256
static constexpr size_t OFF_AH   = 256;                                 // 2*32*1024 f32
static constexpr size_t OFF_AC   = OFF_AH  + 262144;
static constexpr size_t OFF_DH   = OFF_AC  + 262144;
static constexpr size_t OFF_DC   = OFF_DH  + 262144;
static constexpr size_t OFF_CTX  = OFF_DC  + 262144;                    // 32*512 f32
static constexpr size_t OFF_AHH  = OFF_CTX + 65536;                     // 2*32*1024 f16
static constexpr size_t OFF_DHH  = OFF_AHH + 131072;
static constexpr size_t STATE_BYTES = OFF_DHH + 131072;                 // 1376512
static constexpr size_t OFF_WA   = STATE_BYTES;                         // 256*56*512 f16
static constexpr size_t OFF_WD   = OFF_WA  + (size_t)256*kKtA*512*2;    // 256*80*512 f16
static constexpr size_t OFF_PRE  = OFF_WD  + (size_t)256*kKtD*512*2;    // 500*32*256 f16
static constexpr size_t OFF_PMEM = OFF_PRE + (size_t)kTdec*kB*kPre*2;   // 32*400*128 f32

// ---------------- LDS layout (dynamic) ----------------
static constexpr int LDS_XH      = 0;                    // 32*2560 f16 = 163840
static constexpr int LDS_PREACT  = 163840;               // 4096 f32 = 16384 (S2 scratch overlays)
static constexpr int LDS_PERSIST = LDS_PREACT + 16384;   // aw/awc/lconv/ldense/v
static constexpr int SMEM_BYTES  = LDS_PERSIST + (400 + 400 + kNf*62 + kAdim*kNf + kAdim) * 4;

// ============================================================================
// Precompute kernels
// ============================================================================

__global__ void zero_ws_kernel(float* p, int n) {
  int i = blockIdx.x * blockDim.x + threadIdx.x;
  if (i < n) p[i] = 0.f;
}

// Pack [wih | whh] (f32, row-major [4096, Kih]/[4096, Khh]) into f16 WMMA
// B-fragment tiles: tile (nt,kt) holds B[k][n]=W[n][k] for n=nt*16+lane%16,
// k=kt*32+(lane/16)*16+j, laid out [tile][lane][j] (32B contiguous per lane).
__global__ void pack_lstm_kernel(const float* __restrict__ wih,
                                 const float* __restrict__ whh,
                                 _Float16* __restrict__ dst,
                                 int Kih, int Khh, int Kt) {
  size_t total = (size_t)256 * Kt * 512;
  for (size_t idx = (size_t)blockIdx.x * blockDim.x + threadIdx.x; idx < total;
       idx += (size_t)gridDim.x * blockDim.x) {
    size_t tile = idx >> 9;
    int e    = (int)(idx & 511);
    int nt   = (int)(tile / Kt);
    int kt   = (int)(tile % Kt);
    int lane = e >> 4, j = e & 15;
    int n = nt * 16 + (lane & 15);
    int k = kt * 32 + (lane >> 4) * 16 + j;
    float v = (k < Kih) ? wih[(size_t)n * Kih + k]
                        : whh[(size_t)n * Khh + (k - Kih)];
    dst[idx] = (_Float16)v;
  }
}

// Prenet: pre[t][b][:] = relu(x @ w1.T) @ w2.T, x = (t==0 ? 0 : dec[:,:,t-1])
__global__ void prenet_kernel(const float* __restrict__ dec_in,
                              const float* __restrict__ w1,
                              const float* __restrict__ w2,
                              _Float16* __restrict__ pre) {
  __shared__ float xs[kB * kMel];
  __shared__ float hs[kB * kPre];
  int t = blockIdx.x;
  for (int i = threadIdx.x; i < kB * kMel; i += kThr) {
    int b = i / kMel, m = i % kMel;
    xs[i] = (t == 0) ? 0.f : dec_in[((size_t)b * kMel + m) * kTdec + (t - 1)];
  }
  __syncthreads();
  for (int i = threadIdx.x; i < kB * kPre; i += kThr) {
    int b = i >> 8, j = i & 255;
    float s = 0.f;
    for (int m = 0; m < kMel; ++m) s += xs[b * kMel + m] * w1[j * kMel + m];
    hs[i] = fmaxf(s, 0.f);
  }
  __syncthreads();
  for (int i = threadIdx.x; i < kB * kPre; i += kThr) {
    int b = i >> 8, j = i & 255;
    float s = 0.f;
    for (int k = 0; k < kPre; ++k) s += hs[b * kPre + k] * w2[j * kPre + k];
    pre[(size_t)t * (kB * kPre) + i] = (_Float16)s;
  }
}

// pmem[b][t][:] = memory[b][t][:] @ wmem.T   (K=512 -> 128)
__global__ void pmem_kernel(const float* __restrict__ memory,
                            const float* __restrict__ wmem,
                            float* __restrict__ pmem) {
  __shared__ float row[kEnc];
  int bt = blockIdx.x;  // b*400 + t
  for (int i = threadIdx.x; i < kEnc; i += 128) row[i] = memory[(size_t)bt * kEnc + i];
  __syncthreads();
  int a = threadIdx.x;  // 128 threads
  float s = 0.f;
  for (int k = 0; k < kEnc; ++k) s += row[k] * wmem[a * kEnc + k];
  pmem[(size_t)bt * kAdim + a] = s;
}

// ============================================================================
// Persistent decoder kernel
// ============================================================================

__device__ __forceinline__ void gsync(unsigned* cnt, unsigned target) {
  __syncthreads();
  if (threadIdx.x == 0) {
    __threadfence();                    // publish prior writes device-scope
    atomicAdd(cnt, 1u);
    while (__atomic_load_n((volatile unsigned*)cnt, __ATOMIC_ACQUIRE) < target)
      __builtin_amdgcn_s_sleep(1);
    __threadfence();
  }
  __syncthreads();
}

__device__ __forceinline__ float blk_reduce(float v, float* red, int tid, bool ismax) {
  red[tid] = v;
  __syncthreads();
  for (int s = 128; s > 0; s >>= 1) {
    if (tid < s) red[tid] = ismax ? fmaxf(red[tid], red[tid + s]) : (red[tid] + red[tid + s]);
    __syncthreads();
  }
  float r = red[0];
  __syncthreads();
  return r;
}

__device__ __forceinline__ float sigm(float x) { return 1.f / (1.f + __expf(-x)); }

// One LSTM stage: g = [xh] @ Wpack.T (WMMA), then fused gate nonlinearity.
// 8 waves x 2 tasks = 16 single-gate 16x16 tiles per block; this block owns
// unit tiles u = blockIdx.x*2 + {0,1} (32 hidden units) for all 32 rows.
__device__ __forceinline__ void lstm_stage(
    const _Float16* __restrict__ wpack, int Kt,
    const _Float16* xh_s, float* preact_s,
    const float* __restrict__ bih, const float* __restrict__ bhh,
    const float* __restrict__ c_old, float* __restrict__ c_new,
    float* __restrict__ h_new_f, _Float16* __restrict__ h_new_h,
    int lane, int tid) {
  const int K = Kt * 32;
  const int w = tid >> 5;
  const int hid = lane >> 4;
  for (int c = 0; c < 2; ++c) {
    const int t16 = w * 2 + c;            // = ul*8 + mt*4 + q
    const int q  = t16 & 3;
    const int mt = (t16 >> 2) & 1;
    const int ul = t16 >> 3;
    const int u  = blockIdx.x * 2 + ul;   // unit tile 0..63
    const int nt = q * 64 + u;            // gate-row tile 0..255
    v8f acc;
    #pragma unroll
    for (int i = 0; i < 8; ++i) acc[i] = 0.f;
    const _Float16* arow = xh_s + (size_t)(mt * 16 + (lane & 15)) * K;
    const _Float16* wb   = wpack + (size_t)nt * Kt * 512 + lane * 16;
    for (int kt = 0; kt < Kt; ++kt) {
      union { v16h h; uint4 u4[2]; } a;
      const uint4* src = (const uint4*)(arow + kt * 32);
      a.u4[0] = src[2 * hid];     // K = hid*16 .. +7    -> frag elems 0..7
      a.u4[1] = src[2 * hid + 1]; // K = hid*16+8 .. +7  -> frag elems 8..15
      v16h bf = *(const v16h*)(wb + (size_t)kt * 512);
      if (kt + 4 < Kt)
        __builtin_prefetch((const void*)(wb + (size_t)(kt + 4) * 512), 0, 0);
      acc = __builtin_amdgcn_wmma_f32_16x16x32_f16(
          false, a.h, false, bf, (short)0, acc, false, false);
    }
    float* dst = preact_s + (size_t)t16 * 256;
    #pragma unroll
    for (int r = 0; r < 8; ++r)
      dst[(r + 8 * hid) * 16 + (lane & 15)] = acc[r];   // C layout: m=r+8*half, n=lane%16
  }
  __syncthreads();
  // fused LSTM nonlinearity over this block's 32 units x 32 rows
  for (int e = tid; e < 1024; e += kThr) {
    int m  = e >> 5;              // batch row
    int nl = e & 31;              // local unit
    int ul = nl >> 4, n16 = nl & 15;
    int mt = m >> 4,  ml  = m & 15;
    int n  = blockIdx.x * 32 + nl;  // global hidden unit
    int base = ml * 16 + n16;
    float gi = preact_s[(ul * 8 + mt * 4 + 0) * 256 + base] + bih[n]            + bhh[n];
    float gf = preact_s[(ul * 8 + mt * 4 + 1) * 256 + base] + bih[kRnn + n]     + bhh[kRnn + n];
    float gg = preact_s[(ul * 8 + mt * 4 + 2) * 256 + base] + bih[2 * kRnn + n] + bhh[2 * kRnn + n];
    float go = preact_s[(ul * 8 + mt * 4 + 3) * 256 + base] + bih[3 * kRnn + n] + bhh[3 * kRnn + n];
    float cn = sigm(gf) * c_old[m * kRnn + n] + sigm(gi) * tanhf(gg);
    float hv = sigm(go) * tanhf(cn);
    c_new[m * kRnn + n]   = cn;
    h_new_f[m * kRnn + n] = hv;
    h_new_h[m * kRnn + n] = (_Float16)hv;
  }
}

__global__ __launch_bounds__(kThr, 1) void decoder_main(
    const float* __restrict__ memoryp, const _Float16* __restrict__ preB,
    const float* __restrict__ pmemB,
    const _Float16* __restrict__ wA, const _Float16* __restrict__ wD,
    const float* __restrict__ a_bih, const float* __restrict__ a_bhh,
    const float* __restrict__ d_bih, const float* __restrict__ d_bhh,
    const float* __restrict__ wq_w, const float* __restrict__ v_w_p,
    const float* __restrict__ lconv, const float* __restrict__ ldense,
    const float* __restrict__ projw, const float* __restrict__ projb,
    const float* __restrict__ gatew, const float* __restrict__ gateb,
    const int* __restrict__ memlen,
    float* ah, float* ac, float* dh, float* dc, float* ctxg,
    _Float16* ah_h, _Float16* dh_h, unsigned* cnt,
    float* mel_out, float* gate_out, float* align_out) {
  extern __shared__ char smem[];
  _Float16* xh_s    = (_Float16*)(smem + LDS_XH);
  float*    preact_s = (float*)(smem + LDS_PREACT);
  // stage-2 scratch overlays the preact region (disjoint in time)
  float* pq_s    = preact_s;            // 128
  float* e_s     = preact_s + 128;      // 400
  float* alpha_s = preact_s + 528;      // 400
  float* ctx_s   = preact_s + 928;      // 512
  float* red_s   = preact_s + 1440;     // 256
  // persistent region (lives across all 500 steps)
  float* aw_s     = (float*)(smem + LDS_PERSIST);     // 400: alpha(t-1)
  float* awc_s    = aw_s + 400;                       // 400: cumulative
  float* lconv_s  = awc_s + 400;                      // 32*62
  float* ldense_s = lconv_s + kNf * 62;               // 128*32
  float* v_s      = ldense_s + kAdim * kNf;           // 128

  const int tid  = threadIdx.x;
  const int lane = tid & 31;
  const int b    = blockIdx.x;          // stage-2 batch row owned by this WG

  for (int i = tid; i < kTenc; i += kThr) { aw_s[i] = 0.f; awc_s[i] = 0.f; }
  for (int i = tid; i < kNf * 62; i += kThr) lconv_s[i] = lconv[i];
  for (int i = tid; i < kAdim * kNf; i += kThr) ldense_s[i] = ldense[i];
  if (tid < kAdim) v_s[tid] = v_w_p[tid];
  __syncthreads();

  #pragma unroll 1
  for (int t = 0; t < kTdec; ++t) {
    const int pA = t & 1;       // read parity
    const int pB = pA ^ 1;      // write parity

    // ======== Stage 1: attention LSTM ========
    {  // stage xh = [pre_t(256) | ctx(512) | ah_prev(1024)] as f16 in LDS
      const int br = tid >> 3, sub = tid & 7;
      const _Float16* prow = preB + ((size_t)t * kB + br) * kPre;
      const float*    crow = ctxg + br * kEnc;
      const _Float16* hrow = ah_h + pA * 32768 + br * kRnn;
      _Float16* dst = xh_s + (size_t)br * kKA;
      for (int k = sub; k < kKA; k += 8) {
        _Float16 v;
        if (k < kPre)             v = prow[k];
        else if (k < kPre + kEnc) v = (_Float16)crow[k - kPre];
        else                      v = hrow[k - kPre - kEnc];
        dst[k] = v;
      }
    }
    __syncthreads();
    lstm_stage(wA, kKtA, xh_s, preact_s, a_bih, a_bhh,
               ac + pA * 32768, ac + pB * 32768,
               ah + pB * 32768, ah_h + pB * 32768, lane, tid);
    gsync(cnt, (unsigned)(t * 3 + 1) * kNWG);

    // ======== Stage 2: location-sensitive attention (+ mel/gate proj) ========
    {
      const float* ahf = ah + pB * 32768 + b * kRnn;
      for (int a = tid; a < kAdim; a += kThr) {         // pq = ah @ wq.T
        float s = 0.f;
        const float* wr = wq_w + a * kRnn;
        for (int k = 0; k < kRnn; ++k) s += ahf[k] * wr[k];
        pq_s[a] = s;
      }
      __syncthreads();
      const int mlen = memlen[b];
      for (int tt = tid; tt < kTenc; tt += kThr) {      // conv + dense + energy
        float cv[kNf];
        #pragma unroll
        for (int f = 0; f < kNf; ++f) cv[f] = 0.f;
        for (int tap = 0; tap < 31; ++tap) {
          int p = tt + tap - 15;
          float a0 = 0.f, a1 = 0.f;
          if (p >= 0 && p < kTenc) { a0 = aw_s[p]; a1 = awc_s[p]; }
          #pragma unroll
          for (int f = 0; f < kNf; ++f)
            cv[f] += a0 * lconv_s[f * 62 + tap] + a1 * lconv_s[f * 62 + 31 + tap];
        }
        const float* pmrow = pmemB + ((size_t)b * kTenc + tt) * kAdim;
        float ev = 0.f;
        for (int a = 0; a < kAdim; ++a) {
          float loc = 0.f;
          #pragma unroll
          for (int f = 0; f < kNf; ++f) loc += cv[f] * ldense_s[a * kNf + f];
          ev += tanhf(pq_s[a] + loc + pmrow[a]) * v_s[a];
        }
        e_s[tt] = (tt >= mlen) ? -1e9f : ev;
      }
      __syncthreads();
      // softmax(e)
      float lm = -3.4e38f;
      for (int tt = tid; tt < kTenc; tt += kThr) lm = fmaxf(lm, e_s[tt]);
      lm = blk_reduce(lm, red_s, tid, true);
      float ls = 0.f;
      for (int tt = tid; tt < kTenc; tt += kThr) {
        float ex = __expf(e_s[tt] - lm); alpha_s[tt] = ex; ls += ex;
      }
      ls = blk_reduce(ls, red_s, tid, false);
      float inv = 1.f / ls;
      for (int tt = tid; tt < kTenc; tt += kThr) alpha_s[tt] *= inv;
      __syncthreads();
      if (t > 0) {  // forward attention recursion
        float ps = 0.f;
        for (int tt = tid; tt < kTenc; tt += kThr) {
          float tilde = (tt == 0) ? 0.f : aw_s[tt - 1];
          float ap = alpha_s[tt] * tilde;
          e_s[tt] = ap; ps += ap;
        }
        ps = blk_reduce(ps, red_s, tid, false);
        float rs = 1.f / (ps + 1e-8f);
        float m2 = -3.4e38f;
        for (int tt = tid; tt < kTenc; tt += kThr) {
          float fa = e_s[tt] * rs + 1e-8f; e_s[tt] = fa; m2 = fmaxf(m2, fa);
        }
        m2 = blk_reduce(m2, red_s, tid, true);
        float s2 = 0.f;
        for (int tt = tid; tt < kTenc; tt += kThr) {
          float ex = __expf(e_s[tt] - m2); e_s[tt] = ex; s2 += ex;
        }
        s2 = blk_reduce(s2, red_s, tid, false);
        float i2 = 1.f / s2;
        for (int tt = tid; tt < kTenc; tt += kThr) alpha_s[tt] = e_s[tt] * i2;
        __syncthreads();
      }
      for (int tt = tid; tt < kTenc; tt += kThr) {
        float al = alpha_s[tt];
        aw_s[tt] = al;
        awc_s[tt] += al;
        align_out[((size_t)b * kTdec + t) * kTenc + tt] = al;
      }
      __syncthreads();
      for (int c = tid; c < kEnc; c += kThr) {          // ctx = alpha @ memory
        float s = 0.f;
        const float* mcol = memoryp + (size_t)b * kTenc * kEnc + c;
        for (int tt = 0; tt < kTenc; ++tt) s += alpha_s[tt] * mcol[(size_t)tt * kEnc];
        ctx_s[c] = s;
        ctxg[b * kEnc + c] = s;
      }
      __syncthreads();
      if (tid <= kMel) {  // mel (80) + gate (1) projections from din=[ah|ctx]
        const float* wr = (tid < kMel) ? (projw + tid * (kRnn + kEnc)) : gatew;
        float s = (tid < kMel) ? projb[tid] : gateb[0];
        for (int k = 0; k < kRnn; ++k) s += ahf[k] * wr[k];
        for (int k = 0; k < kEnc; ++k) s += ctx_s[k] * wr[kRnn + k];
        if (tid < kMel) mel_out[((size_t)b * kMel + tid) * kTdec + t] = s;
        else            gate_out[(size_t)b * kTdec + t] = s;
      }
    }
    gsync(cnt, (unsigned)(t * 3 + 2) * kNWG);

    // ======== Stage 3: decoder LSTM ========
    {  // stage xh = [ah(1024) | ctx(512) | dh_prev(1024)]
      const int br = tid >> 3, sub = tid & 7;
      const _Float16* hrow = ah_h + pB * 32768 + br * kRnn;
      const float*    crow = ctxg + br * kEnc;
      const _Float16* drow = dh_h + pA * 32768 + br * kRnn;
      _Float16* dst = xh_s + (size_t)br * kKD;
      for (int k = sub; k < kKD; k += 8) {
        _Float16 v;
        if (k < kRnn)             v = hrow[k];
        else if (k < kRnn + kEnc) v = (_Float16)crow[k - kRnn];
        else                      v = drow[k - kRnn - kEnc];
        dst[k] = v;
      }
    }
    __syncthreads();
    lstm_stage(wD, kKtD, xh_s, preact_s, d_bih, d_bhh,
               dc + pA * 32768, dc + pB * 32768,
               dh + pB * 32768, dh_h + pB * 32768, lane, tid);
    gsync(cnt, (unsigned)(t * 3 + 3) * kNWG);
  }
}

// ============================================================================
extern "C" void kernel_launch(void* const* d_in, const int* in_sizes, int n_in,
                              void* d_out, int out_size, void* d_ws, size_t ws_size,
                              hipStream_t stream) {
  const float* memory = (const float*)d_in[0];
  const float* dec_in = (const float*)d_in[1];
  const float* pw1    = (const float*)d_in[2];
  const float* pw2    = (const float*)d_in[3];
  const float* a_wih  = (const float*)d_in[4];
  const float* a_whh  = (const float*)d_in[5];
  const float* a_bih  = (const float*)d_in[6];
  const float* a_bhh  = (const float*)d_in[7];
  const float* wq     = (const float*)d_in[8];
  const float* wmem   = (const float*)d_in[9];
  const float* v_w    = (const float*)d_in[10];
  const float* lconv  = (const float*)d_in[11];
  const float* ldense = (const float*)d_in[12];
  const float* d_wih  = (const float*)d_in[13];
  const float* d_whh  = (const float*)d_in[14];
  const float* d_bih  = (const float*)d_in[15];
  const float* d_bhh  = (const float*)d_in[16];
  const float* projw  = (const float*)d_in[17];
  const float* projb  = (const float*)d_in[18];
  const float* gatew  = (const float*)d_in[19];
  const float* gateb  = (const float*)d_in[20];
  const int*   memlen = (const int*)d_in[21];

  float* out = (float*)d_out;
  float* mel_out   = out;                                  // [32,80,500]
  float* gate_out  = out + (size_t)kB * kMel * kTdec;      // [32,500]
  float* align_out = gate_out + (size_t)kB * kTdec;        // [32,500,400]

  char* ws = (char*)d_ws;
  unsigned*  cnt   = (unsigned*)(ws + OFF_CNT);
  float*     ah    = (float*)(ws + OFF_AH);
  float*     ac    = (float*)(ws + OFF_AC);
  float*     dh    = (float*)(ws + OFF_DH);
  float*     dc    = (float*)(ws + OFF_DC);
  float*     ctxg  = (float*)(ws + OFF_CTX);
  _Float16*  ah_h  = (_Float16*)(ws + OFF_AHH);
  _Float16*  dh_h  = (_Float16*)(ws + OFF_DHH);
  _Float16*  wA    = (_Float16*)(ws + OFF_WA);
  _Float16*  wD    = (_Float16*)(ws + OFF_WD);
  _Float16*  preB  = (_Float16*)(ws + OFF_PRE);
  float*     pmemB = (float*)(ws + OFF_PMEM);

  // zero state + barrier counter (deterministic every call)
  {
    int nwords = (int)(STATE_BYTES / 4);
    zero_ws_kernel<<<(nwords + 255) / 256, 256, 0, stream>>>((float*)ws, nwords);
  }
  // weight packing (f16 WMMA fragments, L2-resident thereafter)
  pack_lstm_kernel<<<2048, 256, 0, stream>>>(a_wih, a_whh, wA, 768, 1024, kKtA);
  pack_lstm_kernel<<<2048, 256, 0, stream>>>(d_wih, d_whh, wD, 1536, 1024, kKtD);
  // prenet + projected memory
  prenet_kernel<<<kTdec, kThr, 0, stream>>>(dec_in, pw1, pw2, preB);
  pmem_kernel<<<kB * kTenc, 128, 0, stream>>>(memory, wmem, pmemB);

  // persistent recurrence kernel (32 co-resident WGs, ~208KB LDS each)
  (void)hipFuncSetAttribute((const void*)decoder_main,
                            hipFuncAttributeMaxDynamicSharedMemorySize, SMEM_BYTES);
  decoder_main<<<kNWG, kThr, SMEM_BYTES, stream>>>(
      memory, preB, pmemB, wA, wD, a_bih, a_bhh, d_bih, d_bhh,
      wq, v_w, lconv, ldense, projw, projb, gatew, gateb, memlen,
      ah, ac, dh, dc, ctxg, ah_h, dh_h, cnt, mel_out, gate_out, align_out);
}
